// Generator_27539330302059
// MI455X (gfx1250) — matrix operand order repaired and checked by reading, not compiled, not run on hardware
//
#include <hip/hip_runtime.h>
#include <hip/hip_fp16.h>
#include <hip/hip_bf16.h>

// Problem constants (from reference)
#define BB      256
#define ZZ      32
#define HH      64
#define HINN    64
#define NN      20000
#define RWLEN   16
// Logits kernel tiling: N split into 125 chunks of 160 (= 10 WMMA tiles of 16)
#define NCHUNK        125
#define NPERCHUNK     160
#define TILESPERCHUNK 10
#define NTILES        (NN / 16)          // 1250 WMMA row-tiles of W_up
#define FRAG_HALVES   (32 * 16)          // one A fragment: 32 lanes x 16 halves

typedef __attribute__((ext_vector_type(16))) _Float16 v16h;
typedef __attribute__((ext_vector_type(8)))  float    v8f;

__device__ __forceinline__ float sigm_(float v) { return 1.0f / (1.0f + expf(-v)); }

// Counter-based hash -> U(0,1). Deterministic per (t,b,n).
__device__ __forceinline__ float u01_(unsigned x) {
    x ^= x >> 16; x *= 0x7feb352dU;
    x ^= x >> 15; x *= 0x846ca68bU;
    x ^= x >> 16;
    return (float)(x >> 8) * (1.0f / 16777216.0f);
}

// ---------------------------------------------------------------------------
// Zero-fill output (poisoned by harness; one-hot scatter follows).
__global__ void zero_fill_kernel(float4* __restrict__ p, long n4) {
    long i = (long)blockIdx.x * blockDim.x + threadIdx.x;
    long stride = (long)gridDim.x * blockDim.x;
    float4 z = {0.f, 0.f, 0.f, 0.f};
    for (; i < n4; i += stride) p[i] = z;
}

// ---------------------------------------------------------------------------
// One-shot: convert W_up (f32 [N,H]) into pre-swizzled f16 WMMA-A fragments.
// Fragment layout per 16-row tile g and K-half kk:
//   wupF[((g*2)+kk)*512 + lane*16 .. +15]  == lane's v16h A operand
// 16-bit A layout: lanes 0-15 hold K {0..7,16..23}(+kk*32) of row M=lane;
//                  lanes 16-31 hold K {8..15,24..31} of row M=lane-16.
__global__ __launch_bounds__(32) void pack_wup_kernel(
    const float* __restrict__ W_up, _Float16* __restrict__ wupF)
{
    const int g    = blockIdx.x;           // tile (16 rows of W_up)
    const int lane = threadIdx.x;
    const int half = lane >> 4;
    const int l16  = lane & 15;
    const int kh   = half * 8;
    const float* wrow = W_up + (size_t)(g * 16 + l16) * HH;
    #pragma unroll
    for (int kk = 0; kk < 2; ++kk) {
        _Float16* dst = wupF + ((size_t)g * 2 + kk) * FRAG_HALVES + lane * 16;
        v16h a;
        #pragma unroll
        for (int j = 0; j < 8; ++j) {
            a[j]     = (_Float16)wrow[kk * 32 + kh + j];
            a[8 + j] = (_Float16)wrow[kk * 32 + 16 + kh + j];
        }
        *(v16h*)dst = a;
    }
}

// ---------------------------------------------------------------------------
// inter = tanh(latent@W_int.T+b); h0 = tanh(inter@W_hup.T+b); c0 = tanh(inter@W_cup.T+b)
// Also seeds x = inputs and hT (f16, [k][b]) for the WMMA B operand.
__global__ __launch_bounds__(HH) void init_kernel(
    const float* __restrict__ latent, const float* __restrict__ inputs,
    const float* __restrict__ W_int, const float* __restrict__ b_int,
    const float* __restrict__ W_cup, const float* __restrict__ b_cup,
    const float* __restrict__ W_hup, const float* __restrict__ b_hup,
    float* __restrict__ h, float* __restrict__ c, float* __restrict__ x,
    _Float16* __restrict__ hT)
{
    __shared__ float lat_s[ZZ];
    __shared__ float inter_s[HH];
    const int b = blockIdx.x, i = threadIdx.x;
    if (i < ZZ) lat_s[i] = latent[b * ZZ + i];
    __syncthreads();
    float a = b_int[i];
    #pragma unroll
    for (int k = 0; k < ZZ; ++k) a += lat_s[k] * W_int[i * ZZ + k];
    inter_s[i] = tanhf(a);
    __syncthreads();
    float ah = b_hup[i], ac = b_cup[i];
    #pragma unroll 8
    for (int k = 0; k < HH; ++k) {
        const float v = inter_s[k];
        ah += v * W_hup[i * HH + k];
        ac += v * W_cup[i * HH + k];
    }
    const float hv = tanhf(ah), cv = tanhf(ac);
    h[b * HH + i] = hv;
    c[b * HH + i] = cv;
    hT[i * BB + b] = (_Float16)hv;
    x[b * HINN + i] = inputs[b * HINN + i];
}

// ---------------------------------------------------------------------------
// gates = x@W_ih.T + b_ih + h@W_hh.T + b_hh ; LSTM cell update.
// One block per batch row; 256 threads compute the 4H gates, 64 do the update.
__global__ __launch_bounds__(4 * HH) void lstm_step_kernel(
    const float* __restrict__ W_ih, const float* __restrict__ b_ih,
    const float* __restrict__ W_hh, const float* __restrict__ b_hh,
    float* __restrict__ h, float* __restrict__ c, const float* __restrict__ x,
    _Float16* __restrict__ hT)
{
    __shared__ float xs[HINN];
    __shared__ float hs[HH];
    __shared__ float gs[4 * HH];
    const int b = blockIdx.x, j = threadIdx.x;
    if (j < HINN) xs[j] = x[b * HINN + j];
    else if (j < HINN + HH) hs[j - HINN] = h[b * HH + (j - HINN)];
    __syncthreads();
    float a = b_ih[j] + b_hh[j];
    const float* wi = W_ih + (size_t)j * HINN;
    const float* wh = W_hh + (size_t)j * HH;
    #pragma unroll 8
    for (int k = 0; k < HINN; ++k) a += xs[k] * wi[k];
    #pragma unroll 8
    for (int k = 0; k < HH; ++k) a += hs[k] * wh[k];
    gs[j] = a;
    __syncthreads();
    if (j < HH) {
        const float iv = sigm_(gs[j]);
        const float fv = sigm_(gs[HH + j]);
        const float gv = tanhf(gs[2 * HH + j]);
        const float ov = sigm_(gs[3 * HH + j]);
        const float cv = fv * c[b * HH + j] + iv * gv;
        const float hv = ov * tanhf(cv);
        c[b * HH + j] = cv;
        h[b * HH + j] = hv;
        hT[j * BB + b] = (_Float16)hv;   // f16 transposed copy for WMMA B operand
    }
}

// ---------------------------------------------------------------------------
// logits^T tile GEMM via v_wmma_f32_16x16x32_f16, fused gumbel + running argmax.
//   A (16xK)  = pre-swizzled f16 fragment of W_up (two b128 loads per K-half)
//   B (Kx16)  = h^T for a 16-batch tile (from hT[k][b])
//   C (16x16) = logits^T : lane holds 8 N-rows of ONE batch column
// One wave per block; grid = (NCHUNK, B/16). K=64 -> two WMMAs per tile.
__global__ __launch_bounds__(32) void logits_argmax_kernel(
    const _Float16* __restrict__ wupF, const float* __restrict__ b_up,
    const _Float16* __restrict__ hT,
    float* __restrict__ pV, int* __restrict__ pI, int t)
{
    const int lane = threadIdx.x;          // 0..31, full wave, EXEC all ones
    const int half = lane >> 4;            // 0: lanes 0-15, 1: lanes 16-31
    const int l16  = lane & 15;
    const int chunk  = blockIdx.x;         // N chunk: 10 row-tiles of logits^T
    const int b_base = blockIdx.y * 16;    // batch tile
    const int bidx   = b_base + l16;       // this lane's batch column
    const int g0     = chunk * TILESPERCHUNK;   // first W_up tile of this chunk

    // B operand: h^T, lanes 0-15 hold K {0..7,16..23}, lanes 16-31 hold K {8..15,24..31}
    v16h bm0, bm1;
    {
        const int kb0 = half * 8;
        const int kb1 = 32 + half * 8;
        #pragma unroll
        for (int j = 0; j < 16; ++j) bm0[j] = hT[(kb0 + (j & 7) + ((j >> 3) << 4)) * BB + bidx];
        #pragma unroll
        for (int j = 0; j < 16; ++j) bm1[j] = hT[(kb1 + (j & 7) + ((j >> 3) << 4)) * BB + bidx];
    }

    float bestV = -3.4e38f;
    int   bestN = 0;

    for (int tile = 0; tile < TILESPERCHUNK; ++tile) {
        const int g      = g0 + tile;
        const int n_base = g * 16;
        const _Float16* frag = wupF + (size_t)g * 2 * FRAG_HALVES + lane * 16;
        // Prefetch next tile's fragments into cache (global_prefetch_b8)
        if (tile + 1 < TILESPERCHUNK)
            __builtin_prefetch(frag + 2 * FRAG_HALVES, 0, 3);
        const v16h a0 = *(const v16h*)(frag);
        const v16h a1 = *(const v16h*)(frag + FRAG_HALVES);
        v8f acc = {};
        acc = __builtin_amdgcn_wmma_f32_16x16x32_f16(false, a0, false, bm0,
                                                     (short)0, acc, false, false);
        acc = __builtin_amdgcn_wmma_f32_16x16x32_f16(false, a1, false, bm1,
                                                     (short)0, acc, false, false);
        // This lane's 8 consecutive N rows share two float4 loads of b_up
        const float4* bup = (const float4*)(b_up + n_base + half * 8);
        const float4 u0 = bup[0], u1 = bup[1];
        const float bias[8] = {u0.x, u0.y, u0.z, u0.w, u1.x, u1.y, u1.z, u1.w};
        // C layout: VGPR r -> row M = r (lanes 0-15) / 8+r (lanes 16-31), col = lane%16
        #pragma unroll
        for (int r = 0; r < 8; ++r) {
            const int n = n_base + half * 8 + r;
            const unsigned ctr = (unsigned)((t * BB + bidx) * (unsigned)NN + (unsigned)n);
            const float u = u01_(ctr * 0x9E3779B9u + 0x85EBCA6Bu);
            const float gnoise = -logf(-logf(u + 1e-20f) + 1e-20f);   // gumbel, TEMP=1
            const float v = acc[r] + bias[r] + gnoise;
            if (v > bestV) { bestV = v; bestN = n; }
        }
    }
    // lanes l and l+16 hold the same batch column: combine halves
    const float oV = __shfl_xor(bestV, 16, 32);
    const int   oN = __shfl_xor(bestN, 16, 32);
    if (oV > bestV || (oV == bestV && oN < bestN)) { bestV = oV; bestN = oN; }
    if (half == 0) {
        pV[bidx * NCHUNK + chunk] = bestV;
        pI[bidx * NCHUNK + chunk] = bestN;
    }
}

// ---------------------------------------------------------------------------
// Final argmax over 125 chunk-partials per batch; scatter one-hot into d_out;
// x = W_down[:, argmax] + b_down  (next step input).
__global__ __launch_bounds__(128) void finalize_kernel(
    const float* __restrict__ pV, const int* __restrict__ pI,
    const float* __restrict__ W_down, const float* __restrict__ b_down,
    float* __restrict__ x, float* __restrict__ out, int t)
{
    __shared__ float sv[128];
    __shared__ int   si[128];
    __shared__ int   sidx;
    const int b = blockIdx.x, tid = threadIdx.x;
    float bv = -3.4e38f; int bi = 0;
    for (int cidx = tid; cidx < NCHUNK; cidx += 128) {
        const float v = pV[b * NCHUNK + cidx];
        if (v > bv) { bv = v; bi = pI[b * NCHUNK + cidx]; }
    }
    sv[tid] = bv; si[tid] = bi;
    __syncthreads();
    for (int s = 64; s > 0; s >>= 1) {
        if (tid < s && sv[tid + s] > sv[tid]) { sv[tid] = sv[tid + s]; si[tid] = si[tid + s]; }
        __syncthreads();
    }
    if (tid == 0) {
        sidx = si[0];
        out[(size_t)b * RWLEN * NN + (size_t)t * NN + si[0]] = 1.0f;   // one-hot
    }
    __syncthreads();
    if (tid < HINN)
        x[b * HINN + tid] = W_down[(size_t)tid * NN + sidx] + b_down[tid];
}

// ---------------------------------------------------------------------------
extern "C" void kernel_launch(void* const* d_in, const int* in_sizes, int n_in,
                              void* d_out, int out_size, void* d_ws, size_t ws_size,
                              hipStream_t stream) {
    const float* latent = (const float*)d_in[0];
    const float* inputs = (const float*)d_in[1];
    const float* W_int  = (const float*)d_in[2];
    const float* b_int  = (const float*)d_in[3];
    const float* W_cup  = (const float*)d_in[4];
    const float* b_cup  = (const float*)d_in[5];
    const float* W_hup  = (const float*)d_in[6];
    const float* b_hup  = (const float*)d_in[7];
    const float* W_ih   = (const float*)d_in[8];
    const float* b_ih   = (const float*)d_in[9];
    const float* W_hh   = (const float*)d_in[10];
    const float* b_hh   = (const float*)d_in[11];
    const float* W_up   = (const float*)d_in[12];
    const float* b_up   = (const float*)d_in[13];
    const float* W_down = (const float*)d_in[14];
    const float* b_down = (const float*)d_in[15];
    float* out = (float*)d_out;

    // Workspace carve-up (all offsets 128B-aligned)
    char* ws = (char*)d_ws;
    float*    h    = (float*)(ws);                         // 256*64*4  = 64 KB
    float*    c    = (float*)(ws + 65536);                 // 64 KB
    float*    x    = (float*)(ws + 131072);                // 64 KB
    _Float16* hT   = (_Float16*)(ws + 196608);             // 64*256*2  = 32 KB
    float*    pV   = (float*)(ws + 229376);                // 256*125*4 = 125 KB
    int*      pI   = (int*)(ws + 357376);                  // 125 KB
    _Float16* wupF = (_Float16*)(ws + 485376);             // 20000*64*2 = 2.56 MB

    // 1) zero the 327 MB one-hot output (memory-bound: ~14 us @ 23.3 TB/s)
    {
        const long n4 = (long)out_size / 4;                // out_size % 4 == 0
        zero_fill_kernel<<<4096, 256, 0, stream>>>((float4*)out, n4);
    }
    // 2) one-shot W_up f32 -> pre-swizzled f16 A-fragments (L2-resident, reused 16x)
    pack_wup_kernel<<<NTILES, 32, 0, stream>>>(W_up, wupF);
    // 3) initial state
    init_kernel<<<BB, HH, 0, stream>>>(latent, inputs, W_int, b_int, W_cup, b_cup,
                                       W_hup, b_hup, h, c, x, hT);
    // 4) 16 sequential walk steps
    for (int t = 0; t < RWLEN; ++t) {
        lstm_step_kernel<<<BB, 4 * HH, 0, stream>>>(W_ih, b_ih, W_hh, b_hh, h, c, x, hT);
        logits_argmax_kernel<<<dim3(NCHUNK, BB / 16), 32, 0, stream>>>(
            wupF, b_up, hT, pV, pI, t);
        finalize_kernel<<<BB, 128, 0, stream>>>(pV, pI, W_down, b_down, x, out, t);
    }
    (void)in_sizes; (void)n_in; (void)ws_size;
}
